// GRU_55740085567738
// MI455X (gfx1250) — compile-verified, matrix operand order checked by reading, not run on hardware
//
#include <hip/hip_runtime.h>
#include <hip/hip_bf16.h>

// GRU: T=1024, B=64, I=H=256.
// Kernel 1: input projections Xf/Xi/Xu = X @ W_{f,i,u}x via f16 WMMA (HBM-bound).
// Kernel 2: sequential scan split into 4 independent batch streams (batch rows
//           are independent through the recurrence): 4 workgroups x 16 waves,
//           each block owns 16 batch rows (one WMMA M-tile). Recurrent weights
//           are register-resident WMMA-B fragments; h lives in LDS in WMMA-A
//           fragment-major f16 layout. Per-step gate inputs are streamed into
//           LDS with GLOBAL_LOAD_ASYNC_TO_LDS (ASYNCcnt) one step ahead.

typedef _Float16 v16h __attribute__((ext_vector_type(16)));
typedef _Float16 v4h  __attribute__((ext_vector_type(4)));
typedef float    v8f  __attribute__((ext_vector_type(8)));

#define HDIM 256
#define BDIM 64
#define TDIM 1024

#if defined(__has_builtin)
#  if __has_builtin(__builtin_amdgcn_global_load_async_to_lds_b128)
#    define GRU_HAVE_ASYNC 1
#  endif
#endif
#ifndef GRU_HAVE_ASYNC
#  define GRU_HAVE_ASYNC 0
#endif

#if GRU_HAVE_ASYNC
// Diagnostic from the probe round shows param 0 is
// '__attribute__((__vector_size__(4*sizeof(int)))) int __device__ *'
// i.e. a non-const 16-byte int vector pointer in the global (AS1) address
// space; the LDS-side param mirrors it in AS3 (__shared__).
typedef int v4i_t __attribute__((vector_size(16)));
typedef __attribute__((address_space(1))) v4i_t g_v4i_t;
typedef __attribute__((address_space(3))) v4i_t l_v4i_t;

static __device__ __forceinline__ void async_ldg_b128(const float* g, float* l) {
  // 16 bytes per lane, memory -> LDS, tracked by ASYNCcnt.
  __builtin_amdgcn_global_load_async_to_lds_b128((g_v4i_t*)g, (l_v4i_t*)l, 0, 0);
}

static __device__ __forceinline__ void async_wait0() {
#  if __has_builtin(__builtin_amdgcn_s_wait_asynccnt)
  __builtin_amdgcn_s_wait_asynccnt(0);
#  else
  asm volatile("s_wait_asynccnt 0x0" ::: "memory");
#  endif
}
#endif

static __device__ __forceinline__ v8f wmma16(v16h a, v16h b, v8f c) {
  // (neg_a, A, neg_b, B, c_mod, C, reuse_a, reuse_b)
  return __builtin_amdgcn_wmma_f32_16x16x32_f16(false, a, false, b, (short)0, c,
                                                false, false);
}

static __device__ __forceinline__ float sigmoidf_(float x) {
  return 1.0f / (1.0f + __expf(-x));
}

// ---------------------------------------------------------------------------
// Kernel 1: Y_g[row, n] = sum_k X[row, k] * W_g[k, n]   (rows = T*B = 65536)
// Block: 256 threads = 8 waves (2 M-tiles x 4 N-tiles). Block tile: 32M x 64N.
// ---------------------------------------------------------------------------
__global__ __launch_bounds__(256, 1)
void gru_xproj_kernel(const float* __restrict__ X,
                      const float* __restrict__ Wfx,
                      const float* __restrict__ Wix,
                      const float* __restrict__ Wux,
                      float* __restrict__ Yf,
                      float* __restrict__ Yi,
                      float* __restrict__ Yu) {
  __shared__ __align__(32) _Float16 aF[2 * 8 * 32 * 16];  // 16 KB
  __shared__ __align__(32) _Float16 bF[4 * 8 * 32 * 16];  // 32 KB

  const int tid = threadIdx.x;
  const int R0  = blockIdx.x * 32;   // 2048 blocks in x
  const int N0  = blockIdx.y * 64;   // 4 blocks in y

  // ---- stage A tile (32 rows x K=256), fp32 -> fp16, fragment-major ----
#pragma unroll
  for (int it = 0; it < 8; ++it) {
    int idx = tid + it * 256;
    int r   = idx >> 6;          // 0..31
    int k0  = (idx & 63) << 2;   // 0..252, multiple of 4
    const float4 v =
        *reinterpret_cast<const float4*>(X + (size_t)(R0 + r) * HDIM + k0);
    int mt = r >> 4, rm = r & 15;
    int c = k0 >> 5, kk = k0 & 31;
    int lane = rm + (((kk >> 3) & 1) << 4);
    int j0   = (kk & 7) + ((kk & 16) >> 1);  // 4 consecutive j, same lane
    v4h h4 = {(_Float16)v.x, (_Float16)v.y, (_Float16)v.z, (_Float16)v.w};
    *reinterpret_cast<v4h*>(&aF[(((mt * 8 + c) * 32 + lane) << 4) + j0]) = h4;
  }
  __syncthreads();

  const int l  = tid & 31;
  const int wv = tid >> 5;
  const int mt = wv >> 2;  // 0..1
  const int nt = wv & 3;   // 0..3

  v16h afrag[8];
#pragma unroll
  for (int c = 0; c < 8; ++c)
    afrag[c] = *reinterpret_cast<const v16h*>(&aF[((mt * 8 + c) * 32 + l) << 4]);

  const float* Wg[3] = {Wfx, Wix, Wux};
  float*       Yg[3] = {Yf, Yi, Yu};

  for (int g = 0; g < 3; ++g) {
    __syncthreads();  // protect bF against overwrite while still being read
    const float* W = Wg[g];
#pragma unroll
    for (int it = 0; it < 16; ++it) {
      int idx = tid + it * 256;
      int k   = idx >> 4;          // 0..255
      int cl0 = (idx & 15) << 2;   // local col 0..60
      const float4 w =
          *reinterpret_cast<const float4*>(W + (size_t)k * HDIM + N0 + cl0);
      int c = k >> 5, kk = k & 31;
      int j  = (kk & 7) + ((kk & 16) >> 1);
      int hb = ((kk >> 3) & 1) << 4;
      float ws4[4] = {w.x, w.y, w.z, w.w};
#pragma unroll
      for (int e = 0; e < 4; ++e) {
        int cl = cl0 + e;
        int ntb = cl >> 4, ln = cl & 15;
        bF[(((ntb * 8 + c) * 32 + ln + hb) << 4) + j] = (_Float16)ws4[e];
      }
    }
    __syncthreads();

    v8f acc = {};
#pragma unroll
    for (int c = 0; c < 8; ++c) {
      v16h bfr =
          *reinterpret_cast<const v16h*>(&bF[((nt * 8 + c) * 32 + l) << 4]);
      acc = wmma16(afrag[c], bfr, acc);
    }
    // C layout: m = v + 8*(lane>=16), n = lane&15
    const int colg = N0 + nt * 16 + (l & 15);
    const int rb   = R0 + mt * 16 + ((l >> 4) << 3);
    float* Y = Yg[g];
#pragma unroll
    for (int v = 0; v < 8; ++v)
      Y[(size_t)(rb + v) * HDIM + colg] = acc[v];
  }
}

// ---------------------------------------------------------------------------
// Kernel 2: sequential scan. 4 independent blocks; block b owns batch rows
// [16b, 16b+16). 512 threads = 16 waves; wave n owns output columns
// [16n, 16n+16). Gate inputs for step t+1 are async-DMA'd into LDS while
// step t+1's f/i WMMAs run (wait folded into an existing barrier).
// ---------------------------------------------------------------------------
__global__ __launch_bounds__(512, 1)
void gru_scan_kernel(const float* __restrict__ Wfg, const float* __restrict__ bfg,
                     const float* __restrict__ Wig, const float* __restrict__ big,
                     const float* __restrict__ Wug, const float* __restrict__ bug,
                     const float* __restrict__ Yf, const float* __restrict__ Yi,
                     const float* __restrict__ Yu,
                     float* __restrict__ outs, float* __restrict__ hlast) {
  __shared__ __align__(32) _Float16 hF[8 * 32 * 16];  // 8 KB, A-fragment layout
#if GRU_HAVE_ASYNC
  __shared__ __align__(16) float stageX[3][16][HDIM];  // 48 KB staged x-proj
#endif

  const int tid     = threadIdx.x;
  const int l       = tid & 31;
  const int n       = tid >> 5;     // wave id == N-tile
  const int ln      = l & 15;
  const int col     = n * 16 + ln;  // this lane's output / hidden column
  const int rowbase = blockIdx.x * 16;  // this block's batch rows

  const float* Ys[3] = {Yf, Yi, Yu};

  // ---- load recurrent-weight B fragments into registers (once) ----
  v16h wf[8], wi[8], wu[8];
#pragma unroll
  for (int c = 0; c < 8; ++c) {
    v16h f, i, u;
#pragma unroll
    for (int j = 0; j < 16; ++j) {
      int k = c * 32 + (j & 7) + ((l & 16) >> 1) + ((j & 8) << 1);
      f[j] = (_Float16)(Wfg[(size_t)k * HDIM + col] + bfg[col]);
      i[j] = (_Float16)(Wig[(size_t)k * HDIM + col] + big[col]);
      u[j] = (_Float16)(Wug[(size_t)k * HDIM + col] + bug[col]);
    }
    wf[c] = f; wi[c] = i; wu[c] = u;
  }

  // ---- h0 = 0 ----
  for (int idx = tid; idx < 8 * 32 * 16; idx += 512) hF[idx] = (_Float16)0.0f;
  v8f hv = {};

#if GRU_HAVE_ASYNC
  // stage t = 0 (3 gates x 16 rows x 1 KB; 6 x b128 per thread)
  {
    const size_t tb0 = (size_t)rowbase * HDIM;
#pragma unroll
    for (int g = 0; g < 3; ++g)
#pragma unroll
      for (int q = 0; q < 2; ++q) {
        int flat = tid + q * 512;
        int row = flat >> 6, seg = flat & 63;
        async_ldg_b128(Ys[g] + tb0 + (size_t)row * HDIM + seg * 4,
                       &stageX[g][row][seg * 4]);
      }
  }
  async_wait0();
#endif
  __syncthreads();

  // Scatter constants: this lane's column as a K-index into the A layout.
  const int cM    = col >> 5;
  const int kkM   = col & 31;
  const int jA    = (kkM & 7) + ((kkM & 16) >> 1);
  const int hiSel = (kkM >> 3) & 1;
  const int rmB   = (l >> 4) << 3;  // row-in-tile = rmB + v

  for (int t = 0; t < TDIM; ++t) {
    const size_t base = (size_t)t * BDIM * HDIM + (size_t)rowbase * HDIM;

    // ---- reset & update gates (async DMA for step t still landing) ----
    v8f af = {}, ai = {};
#pragma unroll
    for (int c = 0; c < 8; ++c) {
      v16h a = *reinterpret_cast<const v16h*>(&hF[(c * 32 + l) << 4]);
      af = wmma16(a, wf[c], af);
      ai = wmma16(a, wi[c], ai);
    }
#if GRU_HAVE_ASYNC
    async_wait0();  // this wave's DMA done; barrier makes all waves' visible
#endif
    __syncthreads();  // (1) hF(h) reads complete + staged X[t] visible

    v8f rr, uu;
#pragma unroll
    for (int v = 0; v < 8; ++v) {
#if GRU_HAVE_ASYNC
      float xf = stageX[0][rmB + v][col];
      float xi = stageX[1][rmB + v][col];
#else
      float xf = Yf[base + (size_t)(rmB + v) * HDIM + col];
      float xi = Yi[base + (size_t)(rmB + v) * HDIM + col];
#endif
      rr[v] = sigmoidf_(af[v] + xf);
      uu[v] = sigmoidf_(ai[v] + xi);
    }

    // ---- write reset .* h into hF (A layout; wave owns its K-columns) ----
#pragma unroll
    for (int v = 0; v < 8; ++v) {
      int Alane = rmB + v + (hiSel << 4);
      hF[((cM * 32 + Alane) << 4) + jA] = (_Float16)(rr[v] * hv[v]);
    }
    __syncthreads();  // (2)

    // ---- candidate gate on (reset .* h) ----
    v8f au = {};
#pragma unroll
    for (int c = 0; c < 8; ++c) {
      v16h a = *reinterpret_cast<const v16h*>(&hF[(c * 32 + l) << 4]);
      au = wmma16(a, wu[c], au);
    }
#pragma unroll
    for (int v = 0; v < 8; ++v) {
#if GRU_HAVE_ASYNC
      float xu = stageX[2][rmB + v][col];
#else
      float xu = Yu[base + (size_t)(rmB + v) * HDIM + col];
#endif
      float cand = tanhf(au[v] + xu);
      hv[v]      = uu[v] * hv[v] + (1.0f - uu[v]) * cand;
    }
    __syncthreads();  // (3) hF(rh) reads + stage[t] reads complete

    // ---- publish h_t: LDS (A layout, next step) + global outs[t] ----
#pragma unroll
    for (int v = 0; v < 8; ++v) {
      int Alane = rmB + v + (hiSel << 4);
      hF[((cM * 32 + Alane) << 4) + jA]           = (_Float16)hv[v];
      outs[base + (size_t)(rmB + v) * HDIM + col] = hv[v];
    }
    __syncthreads();  // (4)

    if (t + 1 < TDIM) {
#if GRU_HAVE_ASYNC
      // ---- async DMA step t+1's tile; overlaps next step's f/i WMMAs ----
      const size_t tbn = base + (size_t)BDIM * HDIM;
#pragma unroll
      for (int g = 0; g < 3; ++g)
#pragma unroll
        for (int q = 0; q < 2; ++q) {
          int flat = tid + q * 512;
          int row = flat >> 6, seg = flat & 63;
          async_ldg_b128(Ys[g] + tbn + (size_t)row * HDIM + seg * 4,
                         &stageX[g][row][seg * 4]);
        }
#else
      const size_t nb = base + (size_t)BDIM * HDIM;  // same rows, t+1
      __builtin_prefetch(Yf + nb + (size_t)tid * 8, 0, 1);
      __builtin_prefetch(Yi + nb + (size_t)tid * 8, 0, 1);
      __builtin_prefetch(Yu + nb + (size_t)tid * 8, 0, 1);
#endif
    }
  }

  // ---- h_last ----
#pragma unroll
  for (int v = 0; v < 8; ++v)
    hlast[(size_t)(rowbase + rmB + v) * HDIM + col] = hv[v];
}

extern "C" void kernel_launch(void* const* d_in, const int* in_sizes, int n_in,
                              void* d_out, int out_size, void* d_ws,
                              size_t ws_size, hipStream_t stream) {
  const float* X    = (const float*)d_in[0];
  const float* W_fx = (const float*)d_in[1];
  const float* W_fg = (const float*)d_in[2];
  const float* b_fg = (const float*)d_in[3];
  const float* W_ix = (const float*)d_in[4];
  const float* W_ig = (const float*)d_in[5];
  const float* b_ig = (const float*)d_in[6];
  const float* W_ux = (const float*)d_in[7];
  const float* W_ug = (const float*)d_in[8];
  const float* b_ug = (const float*)d_in[9];

  float* out   = (float*)d_out;  // outs [T*B*H] then h_last [B*H]
  float* hlast = out + (size_t)TDIM * BDIM * HDIM;

  const size_t proj = (size_t)TDIM * BDIM * HDIM;  // 16M floats each
  float* Yf = (float*)d_ws;
  float* Yi = Yf + proj;
  float* Yu = Yi + proj;

  dim3 g1((TDIM * BDIM) / 32, HDIM / 64);  // 2048 x 4 blocks
  gru_xproj_kernel<<<g1, 256, 0, stream>>>(X, W_fx, W_ix, W_ux, Yf, Yi, Yu);

  // 4 independent batch streams of 16 rows each.
  gru_scan_kernel<<<BDIM / 16, 512, 0, stream>>>(W_fg, b_fg, W_ig, b_ig,
                                                 W_ug, b_ug, Yf, Yi, Yu,
                                                 out, hlast);
}
